// MiLoLinear_79929341379025
// MI455X (gfx1250) — matrix operand order repaired and checked by reading, not compiled, not run on hardware
//
#include <hip/hip_runtime.h>
#include <hip/hip_bf16.h>

typedef __attribute__((ext_vector_type(16))) _Float16 v16h;
typedef __attribute__((ext_vector_type(8)))  float    v8f;

#define OUT_F 11008
#define IN_F  4096
#define GQ    704512      // number of groups = OUT_F*IN_F/64
#define M_ALL 512

// ---------------------------------------------------------------------------
// Phase A: x_h = (f16)x  and  T_h = (f16)(x @ V.T)   [512 x 32]
// ---------------------------------------------------------------------------
__global__ __launch_bounds__(256) void milo_phaseA(
    const float* __restrict__ x, const float* __restrict__ V,
    _Float16* __restrict__ x_h, _Float16* __restrict__ T_h)
{
    const int m    = blockIdx.x;
    const int t    = threadIdx.x;
    const int lane = t & 31;
    const int wid  = t >> 5;

    __shared__ float red[32][8];   // [j][wave]

    float partial[32];
    #pragma unroll
    for (int j = 0; j < 32; ++j) partial[j] = 0.f;

    const float* xrow = x + (long)m * IN_F;
    for (int k = t; k < IN_F; k += 256) {
        float xv = xrow[k];
        x_h[(long)m * IN_F + k] = (_Float16)xv;
        #pragma unroll
        for (int j = 0; j < 32; ++j)
            partial[j] += xv * V[(long)j * IN_F + k];
    }

    #pragma unroll
    for (int j = 0; j < 32; ++j) {
        float v = partial[j];
        #pragma unroll
        for (int off = 16; off > 0; off >>= 1)
            v += __shfl_xor(v, off, 32);
        if (lane == 0) red[j][wid] = v;
    }
    __syncthreads();

    if (t < 32) {
        float s = 0.f;
        #pragma unroll
        for (int w = 0; w < 8; ++w) s += red[t][w];
        T_h[(long)m * 32 + t] = (_Float16)s;
    }
}

// ---------------------------------------------------------------------------
// Dequantize one thread's (o_local, 8 k) strip of a 32K x 64N tile into the
// K-major LDS buffer (row stride 80 halves: 16B-aligned, bank-staggered).
// ---------------------------------------------------------------------------
__device__ __forceinline__ void dequant_strip(
    _Float16 (*buf)[80], const int* __restrict__ wq_row,
    const float* __restrict__ scale, const float* __restrict__ zero,
    long cbase, int k0, int kq, int o_local, int qshift)
{
    const long c = cbase + k0;
    #pragma unroll
    for (int j = 0; j < 8; ++j) {
        int   q = (wq_row[c + j] >> qshift) & 0xF;
        float w = ((float)q - zero[c + j]) * scale[c + j];
        buf[kq * 8 + j][o_local] = (_Float16)w;
    }
}

__device__ __forceinline__ v16h load_b_frag(const _Float16* p)
{
    v16h v;
    #pragma unroll
    for (int s = 0; s < 8; ++s) {
        v[s]     = p[s * 80];
        v[s + 8] = p[(s + 16) * 80];
    }
    return v;
}

__device__ __forceinline__ v16h load_a_frag(const _Float16* p)
{
    union { v16h v; uint4 q[2]; } u;
    u.q[0] = *(const uint4*)p;          // K kb .. kb+7
    u.q[1] = *(const uint4*)(p + 16);   // K kb+16 .. kb+23
    return u.v;
}

// ---------------------------------------------------------------------------
// Main: out = x @ W.T + (x@V.T) @ U.T + bias  via v_wmma_f32_16x16x32_f16.
// Block tile 256M x 64N, 8 waves (4x2), each wave 64M x 32N (4x2 fragments).
// Double-buffered LDS weight tile: one barrier per K-step; dequant of step+1
// overlaps the WMMAs of step.
// ---------------------------------------------------------------------------
__global__ __launch_bounds__(256) void milo_main(
    const int*   __restrict__ Wq,   const float* __restrict__ scale,
    const float* __restrict__ zero, const float* __restrict__ U,
    const float* __restrict__ bias, const _Float16* __restrict__ x_h,
    const _Float16* __restrict__ T_h, float* __restrict__ out)
{
    __shared__ _Float16 Wt[2][32][80];

    const int tid  = threadIdx.x;
    const int lane = tid & 31;
    const int wid  = tid >> 5;
    const int wm   = wid >> 1;           // 0..3  (M direction)
    const int wn   = wid & 1;            // 0..1  (N direction)
    const int nblk = blockIdx.x * 64;
    const int mblk = blockIdx.y * 256;
    const int mwave = mblk + wm * 64;
    const int nwave = wn * 32;           // LDS-local N offset of this wave

    // --- dequant mapping: each thread owns (o_local, 8 consecutive k) ---
    const int o_local = tid >> 2;        // 0..63
    const int kq      = tid & 3;         // 0..3 -> k = kq*8 .. +7
    const int o       = nblk + o_local;
    const int r       = o / 172;                       // group row 0..63
    const int om      = o - r * 172;
    const int qrow    = (r < 32) ? r : (r - 32);
    const int qshift  = (r < 32) ? 4 : 0;
    const long cbase  = (long)om * IN_F + kq * 8;      // group index base
    const int* wq_row = Wq + (long)qrow * GQ;

    v8f acc[4][2];
    #pragma unroll
    for (int mi = 0; mi < 4; ++mi)
        #pragma unroll
        for (int ni = 0; ni < 2; ++ni)
            acc[mi][ni] = (v8f){0.f,0.f,0.f,0.f,0.f,0.f,0.f,0.f};

    // Prologue: fill buffer 0 with K-step 0.
    dequant_strip(Wt[0], wq_row, scale, zero, cbase, 0, kq, o_local, qshift);
    __syncthreads();

    const int arow_off = (lane & 15);
    const int akb_off  = (lane >> 4) << 3;

    for (int step = 0; step < 128; ++step) {
        const int cur = step & 1;

        // B fragments from current buffer: lane%16 = N column, 16 halves along K (table order).
        v16h bfrag[2];
        #pragma unroll
        for (int ni = 0; ni < 2; ++ni)
            bfrag[ni] = load_b_frag(&Wt[cur][akb_off][nwave + ni * 16 + arow_off]);

        // Produce next tile into the other buffer (overlaps the WMMAs).
        if (step < 127) {
            if (step < 126)   // speculative prefetch two K-slices ahead
                __builtin_prefetch((const void*)(wq_row + cbase + step * 32 + 64), 0, 1);
            dequant_strip(Wt[cur ^ 1], wq_row, scale, zero,
                          cbase, (step + 1) * 32, kq, o_local, qshift);
        } else {
            // Last slot: stage rank-32 correction  B = U.T into buffer 0.
            const float* up = U + (long)o * 32 + kq * 8;
            #pragma unroll
            for (int j = 0; j < 8; ++j)
                Wt[0][kq * 8 + j][o_local] = (_Float16)up[j];
        }

        // A fragments straight from global f16 (L2-resident) + 8 WMMAs.
        const int k0 = step * 32;
        #pragma unroll
        for (int mi = 0; mi < 4; ++mi) {
            const int row = mwave + mi * 16 + arow_off;
            v16h afrag = load_a_frag(x_h + (long)row * IN_F + k0 + akb_off);
            #pragma unroll
            for (int ni = 0; ni < 2; ++ni)
                acc[mi][ni] = __builtin_amdgcn_wmma_f32_16x16x32_f16(
                    false, afrag, false, bfrag[ni],
                    (short)0, acc[mi][ni], false, false);
        }
        __syncthreads();
    }

    // Rank-32 tail: A = T_h (512x32 f16), B = U.T tile already in buffer 0.
    {
        v16h bfrag[2];
        #pragma unroll
        for (int ni = 0; ni < 2; ++ni)
            bfrag[ni] = load_b_frag(&Wt[0][akb_off][nwave + ni * 16 + arow_off]);
        #pragma unroll
        for (int mi = 0; mi < 4; ++mi) {
            const int row = mwave + mi * 16 + arow_off;
            v16h afrag = load_a_frag(T_h + (long)row * 32 + akb_off);
            #pragma unroll
            for (int ni = 0; ni < 2; ++ni)
                acc[mi][ni] = __builtin_amdgcn_wmma_f32_16x16x32_f16(
                    false, afrag, false, bfrag[ni],
                    (short)0, acc[mi][ni], false, false);
        }
    }

    // Epilogue: bias add + store (C/D layout: VGPR v -> M = v + (lane/16)*8).
    #pragma unroll
    for (int ni = 0; ni < 2; ++ni) {
        const int ocol = nblk + nwave + ni * 16 + (lane & 15);
        const float b  = bias[ocol];
        #pragma unroll
        for (int mi = 0; mi < 4; ++mi) {
            #pragma unroll
            for (int v = 0; v < 8; ++v) {
                const int mrow = mwave + mi * 16 + ((lane >> 4) << 3) + v;
                out[(long)mrow * OUT_F + ocol] = acc[mi][ni][v] + b;
            }
        }
    }
}

// ---------------------------------------------------------------------------
extern "C" void kernel_launch(void* const* d_in, const int* in_sizes, int n_in,
                              void* d_out, int out_size, void* d_ws, size_t ws_size,
                              hipStream_t stream) {
    const float* x     = (const float*)d_in[0];
    const int*   Wq    = (const int*)  d_in[1];
    const float* scale = (const float*)d_in[2];
    const float* zero  = (const float*)d_in[3];
    const float* U     = (const float*)d_in[4];
    const float* V     = (const float*)d_in[5];
    const float* bias  = (const float*)d_in[6];
    float* out = (float*)d_out;

    _Float16* x_h = (_Float16*)d_ws;                    // 512*4096 f16 = 4 MB
    _Float16* T_h = x_h + (size_t)M_ALL * IN_F;         // 512*32 f16

    milo_phaseA<<<dim3(M_ALL), dim3(256), 0, stream>>>(x, V, x_h, T_h);
    milo_main<<<dim3(OUT_F / 64, M_ALL / 256), dim3(256), 0, stream>>>(
        Wq, scale, zero, U, bias, x_h, T_h, out);
}
